// Classifier_36567351558373
// MI455X (gfx1250) — matrix operand order repaired and checked by reading, not compiled, not run on hardware
//
#include <hip/hip_runtime.h>
#include <hip/hip_bf16.h>
#include <math.h>

typedef __attribute__((ext_vector_type(16))) _Float16 v16h;
typedef __attribute__((ext_vector_type(8)))  float    v8f;

#define NBOX   6000
#define MPAD   6016
#define NW     188          // ceil(6000/32)
#define K1     256
#define HDIM   1024
#define N2     32           // 21 cls + 4 reg, padded
#define THRESH 0.7f

// ---------------- gfx1250 async global->LDS staging ----------------

#if defined(__has_builtin)
#if __has_builtin(__builtin_amdgcn_global_load_async_to_lds_b128)
#define HAVE_ASYNC_LDS 1
#endif
#endif
#ifndef HAVE_ASYNC_LDS
#define HAVE_ASYNC_LDS 0
#endif

// builtin parameter type per hipcc diagnostic:
//   __attribute__((__vector_size__(4*sizeof(int)))) int __device__ *
typedef int v4i_raw __attribute__((vector_size(4 * sizeof(int))));
typedef __attribute__((address_space(1))) v4i_raw g_v4i;   // global src
typedef __attribute__((address_space(3))) v4i_raw l_v4i;   // LDS dst

__device__ __forceinline__ void stage16(const _Float16* g, _Float16* l) {
#if HAVE_ASYNC_LDS
  // GLOBAL_LOAD_ASYNC_TO_LDS_B128: per-lane 16B memory -> LDS, tracked by ASYNCcnt
  __builtin_amdgcn_global_load_async_to_lds_b128(
      (g_v4i*)(unsigned long long)g,
      (l_v4i*)(unsigned int)(unsigned long long)l,
      0, 0);
#else
  *reinterpret_cast<int4*>(l) = *reinterpret_cast<const int4*>(g);
#endif
}

__device__ __forceinline__ void stage_wait() {
#if HAVE_ASYNC_LDS
#if __has_builtin(__builtin_amdgcn_s_wait_asynccnt)
  __builtin_amdgcn_s_wait_asynccnt(0);
#else
  asm volatile("s_wait_asynccnt 0x0" ::: "memory");
#endif
#endif
}

// ---------------- NMS ----------------

__global__ void k_rank(const float* __restrict__ scores, int* __restrict__ order, int n) {
  int i = blockIdx.x * blockDim.x + threadIdx.x;
  if (i >= n) return;
  float si = scores[i];
  int r = 0;
  for (int j = 0; j < n; ++j) {
    float sj = scores[j];
    r += (sj > si) || (sj == si && j < i);
  }
  order[r] = i;   // order = argsort(-scores), stable
}

__global__ void k_gather(const float* __restrict__ boxes, const int* __restrict__ order,
                         float* __restrict__ sb, int n) {
  int i = blockIdx.x * blockDim.x + threadIdx.x;
  if (i >= n) return;
  int o = order[i];
  float x1 = boxes[o * 4 + 0], y1 = boxes[o * 4 + 1];
  float x2 = boxes[o * 4 + 2], y2 = boxes[o * 4 + 3];
  sb[i] = x1; sb[n + i] = y1; sb[2 * n + i] = x2; sb[3 * n + i] = y2;
  sb[4 * n + i] = (x2 - x1) * (y2 - y1);
}

__global__ void k_mask(const float* __restrict__ sb, unsigned* __restrict__ mask,
                       int n, int nw) {
  int t = blockIdx.x * blockDim.x + threadIdx.x;
  if (t >= n * nw) return;
  int i = t / nw, w = t % nw;
  float x1 = sb[i], y1 = sb[n + i], x2 = sb[2 * n + i], y2 = sb[3 * n + i], ai = sb[4 * n + i];
  unsigned bits = 0u;
  int j0 = w * 32;
  for (int b = 0; b < 32; ++b) {
    int j = j0 + b;
    if (j <= i || j >= n) continue;
    float ix1 = fmaxf(x1, sb[j]);
    float iy1 = fmaxf(y1, sb[n + j]);
    float ix2 = fminf(x2, sb[2 * n + j]);
    float iy2 = fminf(y2, sb[3 * n + j]);
    float inter = fmaxf(0.f, ix2 - ix1) * fmaxf(0.f, iy2 - iy1);
    float iou = inter / (ai + sb[4 * n + j] - inter);
    if (iou >= THRESH) bits |= (1u << b);
  }
  mask[(size_t)i * nw + w] = bits;
}

__global__ void k_scan(const unsigned* __restrict__ mask, const int* __restrict__ order,
                       float* __restrict__ keepf, int n, int nw) {
  __shared__ unsigned removed[NW];
  for (int w = threadIdx.x; w < nw; w += blockDim.x) removed[w] = 0u;
  for (int i = 0; i < n; ++i) {
    __syncthreads();   // make prior iteration's updates visible
    bool rem = (removed[i >> 5] >> (i & 31)) & 1u;
    if (threadIdx.x == 0) keepf[order[i]] = rem ? 0.f : 1.f;
    if (!rem) {        // uniform branch (same shared word read by all lanes)
      const unsigned* mi = mask + (size_t)i * nw;
      for (int w = threadIdx.x; w < nw; w += blockDim.x) removed[w] |= mi[w];
    }
  }
}

// ---------------- ROI max-pool -> f16 A matrix ----------------

__global__ void k_roipool(const float* __restrict__ roi, _Float16* __restrict__ flatA,
                          int n, int mpad) {
  int t = blockIdx.x * blockDim.x + threadIdx.x;
  if (t >= mpad * 256) return;
  int nidx = t >> 8, idx = t & 255;
  _Float16 out = (_Float16)0.f;
  if (nidx < n) {
    int i = idx >> 7, j = (idx >> 6) & 1, c = idx & 63;
    const float* base = roi + (((size_t)nidx * 14 + i * 7) * 14 + j * 7) * 64 + c;
    float m = -3.4e38f;
    #pragma unroll
    for (int p = 0; p < 7; ++p)
      #pragma unroll
      for (int q = 0; q < 7; ++q)
        m = fmaxf(m, base[(p * 14 + q) * 64]);
    out = (_Float16)m;
  }
  flatA[(size_t)nidx * 256 + idx] = out;
}

// ---------------- weight prep (n-major f16) ----------------

__global__ void k_w1t(const float* __restrict__ W1, _Float16* __restrict__ Bt) {
  int t = blockIdx.x * blockDim.x + threadIdx.x;
  if (t >= K1 * HDIM) return;
  int k = t >> 10, nn = t & 1023;        // coalesced read of W1[k][n]
  Bt[(size_t)nn * K1 + k] = (_Float16)W1[t];
}

__global__ void k_w2t(const float* __restrict__ Wc, const float* __restrict__ Wr,
                      _Float16* __restrict__ Bt2) {
  int t = blockIdx.x * blockDim.x + threadIdx.x;
  if (t >= N2 * HDIM) return;
  int k = t >> 5, nn = t & 31;
  float v = 0.f;
  if (nn < 21)      v = Wc[k * 21 + nn];
  else if (nn < 25) v = Wr[k * 4 + (nn - 21)];
  Bt2[(size_t)nn * HDIM + k] = (_Float16)v;
}

// ---------------- FC1: (MPADx256) @ (256x1024) via v_wmma_f32_16x16x32_f16 ----
// Whole 64x256 B panel async-staged to LDS once; all A fragments preloaded to
// VGPRs; MAC loop is pure ds_load + wmma with no barriers.

__global__ __launch_bounds__(256) void k_fc1(const _Float16* __restrict__ A,
                                             const _Float16* __restrict__ Bt,
                                             const float* __restrict__ bias,
                                             _Float16* __restrict__ Hout, int mblocks) {
  __shared__ __align__(16) _Float16 Bs[64 * 256];   // 32 KB
  int bx = blockIdx.x;
  int mblk = bx % mblocks;              // 0..46
  int nblk = bx / mblocks;              // 0..15
  int tid = threadIdx.x;
  int wave = tid >> 5, lane = tid & 31;
  int row0 = (mblk * 8 + wave) * 16;
  int col0 = nblk * 64;

  // stage entire B panel: 2048 x 16B chunks, 8 per thread
  #pragma unroll
  for (int r = 0; r < 8; ++r) {
    int c = tid + r * 256;             // 0..2047
    int nn = c >> 5;                   // 0..63  (32 chunks per 256-half row)
    int kp = (c & 31) * 8;             // 0..248
    stage16(Bt + (size_t)(col0 + nn) * K1 + kp, &Bs[nn * 256 + kp]);
  }

  // preload all A fragments for this wave (16 rows x 256 K)
  const int m = lane & 15, kb = (lane >> 4) * 8;    // ISA 16-bit A 16x32 layout
  const _Float16* ap = A + (size_t)(row0 + m) * K1 + kb;
  union Frag { int4 q[2]; v16h v; };
  Frag af[8];
  #pragma unroll
  for (int s = 0; s < 8; ++s) {
    af[s].q[0] = *reinterpret_cast<const int4*>(ap + s * 32);
    af[s].q[1] = *reinterpret_cast<const int4*>(ap + s * 32 + 16);
  }

  stage_wait();
  __syncthreads();

  v8f acc[4] = {};
  const int nB = lane & 15, ksB = (lane >> 4) * 8;  // B fragment lane mapping
  #pragma unroll
  for (int s = 0; s < 8; ++s) {
    int k0 = s * 32;
    #pragma unroll
    for (int tn = 0; tn < 4; ++tn) {
      Frag bf;
      const _Float16* bp = &Bs[(tn * 16 + nB) * 256 + k0 + ksB];
      bf.q[0] = *reinterpret_cast<const int4*>(bp);
      bf.q[1] = *reinterpret_cast<const int4*>(bp + 16);
      acc[tn] = __builtin_amdgcn_wmma_f32_16x16x32_f16(
          false, af[s].v, false, bf.v, (short)0, acc[tn], false, false);
    }
  }

  int lhalf = lane >> 4;
  #pragma unroll
  for (int tn = 0; tn < 4; ++tn) {
    int col = col0 + tn * 16 + (lane & 15);
    float bv = bias[col];
    #pragma unroll
    for (int r = 0; r < 8; ++r) {
      int row = row0 + r + 8 * lhalf;   // C/D layout: VGPR r -> M=r / M=r+8
      float v = acc[tn][r] + bv;
      v = v > 0.f ? v : 0.f;            // ReLU
      Hout[(size_t)row * HDIM + col] = (_Float16)v;
    }
  }
}

// ---------------- FC2: (MPADx1024) @ (1024x32) via WMMA ----------------
// Whole 32x1024 B panel (64 KB) async-staged once; barrier-free MAC loop with
// A prefetch (global_prefetch_b8).

__global__ __launch_bounds__(256) void k_fc2(const _Float16* __restrict__ H,
                                             const _Float16* __restrict__ Bt2,
                                             float* __restrict__ Z) {
  __shared__ __align__(16) _Float16 Bs[32 * 1024];  // 64 KB
  int tid = threadIdx.x, wave = tid >> 5, lane = tid & 31;
  int row0 = (blockIdx.x * 8 + wave) * 16;

  // stage entire B panel: 4096 x 16B chunks, 16 per thread
  #pragma unroll
  for (int r = 0; r < 16; ++r) {
    int c = tid + r * 256;             // 0..4095
    int nn = c >> 7;                   // 0..31  (128 chunks per 1024-half row)
    int kp = (c & 127) * 8;            // 0..1016
    stage16(Bt2 + (size_t)nn * HDIM + kp, &Bs[nn * 1024 + kp]);
  }
  stage_wait();
  __syncthreads();

  v8f acc[2] = {};
  const int m = lane & 15, kb = (lane >> 4) * 8;
  const int nB = lane & 15, ksB = (lane >> 4) * 8;
  const _Float16* ap = H + (size_t)(row0 + m) * HDIM + kb;

  #pragma unroll 4
  for (int k0 = 0; k0 < HDIM; k0 += 32) {
    __builtin_prefetch(ap + k0 + 256, 0, 0);   // global_prefetch_b8 for A stream
    union { int4 q[2]; v16h v; } af;
    af.q[0] = *reinterpret_cast<const int4*>(ap + k0);
    af.q[1] = *reinterpret_cast<const int4*>(ap + k0 + 16);
    #pragma unroll
    for (int tn = 0; tn < 2; ++tn) {
      union { int4 q[2]; v16h v; } bf;
      const _Float16* bp = &Bs[(tn * 16 + nB) * 1024 + k0 + ksB];
      bf.q[0] = *reinterpret_cast<const int4*>(bp);
      bf.q[1] = *reinterpret_cast<const int4*>(bp + 16);
      acc[tn] = __builtin_amdgcn_wmma_f32_16x16x32_f16(
          false, af.v, false, bf.v, (short)0, acc[tn], false, false);
    }
  }

  int lhalf = lane >> 4;
  #pragma unroll
  for (int tn = 0; tn < 2; ++tn) {
    int col = tn * 16 + (lane & 15);
    #pragma unroll
    for (int r = 0; r < 8; ++r) {
      int row = row0 + r + 8 * lhalf;
      Z[(size_t)row * N2 + col] = acc[tn][r];
    }
  }
}

// ---------------- softmax + masking epilogue ----------------

__global__ void k_final(const float* __restrict__ Z, const float* __restrict__ bc,
                        const float* __restrict__ br, const float* __restrict__ keepf,
                        float* __restrict__ out, int n) {
  int i = blockIdx.x * blockDim.x + threadIdx.x;
  if (i >= n) return;
  const float* z = Z + (size_t)i * N2;
  float kf = keepf[i];
  float logits[21];
  float mx = -3.4e38f;
  #pragma unroll
  for (int c = 0; c < 21; ++c) { logits[c] = z[c] + bc[c]; mx = fmaxf(mx, logits[c]); }
  float s = 0.f;
  #pragma unroll
  for (int c = 0; c < 21; ++c) { logits[c] = __expf(logits[c] - mx); s += logits[c]; }
  float inv = kf / s;
  float* cls = out + (size_t)i * 21;
  #pragma unroll
  for (int c = 0; c < 21; ++c) cls[c] = logits[c] * inv;
  float* reg = out + (size_t)NBOX * 21 + (size_t)i * 4;
  #pragma unroll
  for (int r = 0; r < 4; ++r) reg[r] = (z[21 + r] + br[r]) * kf;
  out[(size_t)NBOX * 25 + i] = kf;
}

// ---------------- launch ----------------

extern "C" void kernel_launch(void* const* d_in, const int* in_sizes, int n_in,
                              void* d_out, int out_size, void* d_ws, size_t ws_size,
                              hipStream_t stream) {
  const float* boxes  = (const float*)d_in[0];
  const float* scores = (const float*)d_in[1];
  const float* roi    = (const float*)d_in[2];
  const float* W1     = (const float*)d_in[3];
  const float* b1     = (const float*)d_in[4];
  const float* Wc     = (const float*)d_in[5];
  const float* bc     = (const float*)d_in[6];
  const float* Wr     = (const float*)d_in[7];
  const float* br     = (const float*)d_in[8];
  float* out = (float*)d_out;
  char*  ws  = (char*)d_ws;

  // workspace layout (256B-aligned offsets); NMS mask aliases the h buffer
  const size_t o_keep = 0;                          // 6000 f32
  const size_t o_order = o_keep + 24064;            // 6000 i32
  const size_t o_sbox = o_order + 24064;            // 5*6000 f32
  const size_t o_w1t  = o_sbox + 120064;            // 1024*256 f16
  const size_t o_w2t  = o_w1t + 524288;             // 32*1024 f16
  const size_t o_flat = o_w2t + 65536;              // MPAD*256 f16
  const size_t o_h    = o_flat + 3080192;           // MPAD*1024 f16
  const size_t o_z    = o_h + 12320768;             // MPAD*32 f32
  const size_t o_mask = o_h;                        // 6000*188 u32 (before h is written)

  float*     keepf = (float*)(ws + o_keep);
  int*       order = (int*)(ws + o_order);
  float*     sbox  = (float*)(ws + o_sbox);
  _Float16*  w1t   = (_Float16*)(ws + o_w1t);
  _Float16*  w2t   = (_Float16*)(ws + o_w2t);
  _Float16*  flatA = (_Float16*)(ws + o_flat);
  _Float16*  hbuf  = (_Float16*)(ws + o_h);
  float*     zbuf  = (float*)(ws + o_z);
  unsigned*  mask  = (unsigned*)(ws + o_mask);

  // NMS
  k_rank  <<<(NBOX + 255) / 256, 256, 0, stream>>>(scores, order, NBOX);
  k_gather<<<(NBOX + 255) / 256, 256, 0, stream>>>(boxes, order, sbox, NBOX);
  k_mask  <<<(NBOX * NW + 255) / 256, 256, 0, stream>>>(sbox, mask, NBOX, NW);
  k_scan  <<<1, 256, 0, stream>>>(mask, order, keepf, NBOX, NW);

  // head
  k_roipool<<<(MPAD * 256) / 256, 256, 0, stream>>>(roi, flatA, NBOX, MPAD);
  k_w1t   <<<(K1 * HDIM) / 256, 256, 0, stream>>>(W1, w1t);
  k_w2t   <<<(N2 * HDIM) / 256, 256, 0, stream>>>(Wc, Wr, w2t);

  const int mblocks = MPAD / 128;  // 47
  k_fc1<<<mblocks * (HDIM / 64), 256, 0, stream>>>(flatA, w1t, b1, hbuf, mblocks);
  k_fc2<<<mblocks, 256, 0, stream>>>(hbuf, w2t, zbuf);

  k_final<<<(NBOX + 255) / 256, 256, 0, stream>>>(zbuf, bc, br, keepf, out, NBOX);
}
